// CircleLoss_33827162423499
// MI455X (gfx1250) — matrix operand order repaired and checked by reading, not compiled, not run on hardware
//
#include <hip/hip_runtime.h>
#include <hip/hip_bf16.h>

// CircleLoss fused forward for MI455X (gfx1250, wave32).
// prob tiles computed in-register via v_wmma_f32_16x16x32_f16 (K=64 = 2 WMMAs),
// fused with branchless circle-loss logits + online log-sum-exp reduction.
// Single pass over the upper triangle of sim => HBM-bound (~1.5us @ 23.3 TB/s).

#define NTOT 4096
#define CDIM 64
#define MARGIN_F 0.4f
#define GAMMA_F 80.0f
#define NEG_BIG (-3.402823e38f)

typedef __attribute__((ext_vector_type(16))) _Float16 v16h;
typedef __attribute__((ext_vector_type(8)))  _Float16 v8h;
typedef __attribute__((ext_vector_type(8)))  float    v8f;
typedef __attribute__((ext_vector_type(4)))  float    v4f;

// Branchless online logsumexp accumulate of one weighted term:
//   (m,s) <- combine((m,s), (x, w))   with w in {0,1}
// Safe for m == x == NEG_BIG (w=0 kills the exp(0)=1 artifact).
__device__ __forceinline__ void lse_upd_w(float x, float w, float& m, float& s) {
    float mm = fmaxf(m, x);
    s = s * __expf(m - mm) + w * __expf(x - mm);
    m = mm;
}

__device__ __forceinline__ void lse_merge(float& m, float& s, float mo, float so) {
    float mm = fmaxf(m, mo);
    s = s * __expf(m - mm) + so * __expf(mo - mm);
    m = mm;
}

// ---------------- kernel 1: clusters f32 -> f16 (float4-wide) ----------------
__global__ void __launch_bounds__(256)
circle_cvt_f16(const float* __restrict__ src, _Float16* __restrict__ dst, int n4) {
    int i = blockIdx.x * blockDim.x + threadIdx.x;
    if (i < n4) {
        v4f v = *(const v4f*)(src + (size_t)i * 4);
        _Float16* d = dst + (size_t)i * 4;
        d[0] = (_Float16)v.x; d[1] = (_Float16)v.y;
        d[2] = (_Float16)v.z; d[3] = (_Float16)v.w;
    }
}

// ---------------- kernel 2: fused WMMA + circle loss pass ----------------
// grid.x = 256 row tiles (16 rows each); grid.y = 16 chunks of 16 column tiles.
// 256 threads = 8 wave32; each wave handles up to 2 column tiles of its chunk.
__global__ void __launch_bounds__(256)
circle_main(const float* __restrict__ sim, const _Float16* __restrict__ cl16,
            float* __restrict__ partials) {
    const int bt      = blockIdx.x;            // row-tile index
    const int chunk   = blockIdx.y;            // group of 16 column tiles
    const int wave    = threadIdx.x >> 5;
    const unsigned ln = threadIdx.x & 31u;     // wave32 lane
    const int sel     = (ln >> 4) & 1;         // half-wave select
    const int l16     = ln & 15;

    const int i0 = bt * 16;
    int ctStart = chunk * 16;
    const int ctEnd = ctStart + 16;
    if (ctStart < bt) ctStart = bt;            // strict-upper: tile needed iff ct >= bt

    float m_p = NEG_BIG, s_p = 0.f, m_n = NEG_BIG, s_n = 0.f;
    float cp = 0.f, wp = 0.f, cn = 0.f, wn = 0.f;

    if (ctStart < ctEnd) {
        // A tile (16 x 64 f16), contiguous K to match B:
        // lane<16 holds K {0..15}, lane>=16 holds K {16..31} for its row.
        const _Float16* arow = cl16 + (size_t)(i0 + l16) * CDIM;
        v16h a0, a1;
        {
            v8h c0 = *(const v8h*)(arow      + sel * 16);      // K 0..7   / 16..23
            v8h c1 = *(const v8h*)(arow + 8  + sel * 16);      // K 8..15  / 24..31
            v8h c2 = *(const v8h*)(arow + 32 + sel * 16);      // K 32..39 / 48..55
            v8h c3 = *(const v8h*)(arow + 40 + sel * 16);      // K 40..47 / 56..63
            #pragma unroll
            for (int e = 0; e < 8; ++e) {
                a0[e] = c0[e]; a0[8 + e] = c1[e];
                a1[e] = c2[e]; a1[8 + e] = c3[e];
            }
        }

        for (int ct = ctStart + wave; ct < ctEnd; ct += 8) {
            const int j0 = ct * 16;
            // B tile (32 x 16 f16 per WMMA): lane holds column j0+l16,
            // K 0..15 (lanes 0-15) or K 16..31 (lanes 16-31), contiguous in K.
            // B[k][j] = clusters[j][k]  (prob = clusters @ clusters^T)
            const _Float16* bcol = cl16 + (size_t)(j0 + l16) * CDIM + sel * 16;
            v16h b0 = *(const v16h*)(bcol);        // K 0..31 slice
            v16h b1 = *(const v16h*)(bcol + 32);   // K 32..63 slice

            v8f c = {};
            c = __builtin_amdgcn_wmma_f32_16x16x32_f16(false, a0, false, b0,
                                                       (short)0, c, false, false);
            c = __builtin_amdgcn_wmma_f32_16x16x32_f16(false, a1, false, b1,
                                                       (short)0, c, false, false);

            // C/D layout: VGPR r, lanes 0-15 -> M=r, lanes 16-31 -> M=r+8, N=lane%16
            const int i_base = i0 + sel * 8;
            const int j = j0 + l16;

            // Issue all 8 sim loads up front (clause-friendly, no divergence).
            float sv[8];
            const float* simp = sim + (size_t)i_base * NTOT + j;
            #pragma unroll
            for (int r = 0; r < 8; ++r) sv[r] = simp[(size_t)r * NTOT];

            #pragma unroll
            for (int r = 0; r < 8; ++r) {
                const int i = i_base + r;
                const float s = sv[r];
                const float p = c[r];
                const bool upper = (j > i);
                const bool isP = upper && (p > 0.f);
                const bool isN = upper && !(p > 0.f);
                const float wP = isP ? 1.f : 0.f;
                const float wN = isN ? 1.f : 0.f;

                // logit_p = -relu(1+m-s)*(s-(1-m))*gamma
                const float lp = -fmaxf((1.0f + MARGIN_F) - s, 0.f)
                                 * (s - (1.0f - MARGIN_F)) * GAMMA_F;
                // logit_n =  relu(s+m)*(s-m)*gamma
                const float lnn = fmaxf(s + MARGIN_F, 0.f)
                                  * (s - MARGIN_F) * GAMMA_F;

                const float xp = isP ? lp  : NEG_BIG;
                const float xn = isN ? lnn : NEG_BIG;

                lse_upd_w(xp, wP, m_p, s_p);
                lse_upd_w(xn, wN, m_n, s_n);
                cp += wP; wp += wP * p;
                cn += wN; wn += wN * p;
            }
        }
    }

    // wave32 butterfly reduction
    #pragma unroll
    for (int off = 16; off > 0; off >>= 1) {
        lse_merge(m_p, s_p, __shfl_xor(m_p, off, 32), __shfl_xor(s_p, off, 32));
        lse_merge(m_n, s_n, __shfl_xor(m_n, off, 32), __shfl_xor(s_n, off, 32));
        cp += __shfl_xor(cp, off, 32);
        wp += __shfl_xor(wp, off, 32);
        cn += __shfl_xor(cn, off, 32);
        wn += __shfl_xor(wn, off, 32);
    }

    __shared__ float red[8][8];
    if (ln == 0) {
        float* q = red[wave];
        q[0] = m_p; q[1] = s_p; q[2] = m_n; q[3] = s_n;
        q[4] = cp;  q[5] = wp;  q[6] = cn;  q[7] = wn;
    }
    __syncthreads();
    if (threadIdx.x == 0) {
        #pragma unroll
        for (int w = 1; w < 8; ++w) {
            const float* q = red[w];
            lse_merge(m_p, s_p, q[0], q[1]);
            lse_merge(m_n, s_n, q[2], q[3]);
            cp += q[4]; wp += q[5]; cn += q[6]; wn += q[7];
        }
        float* out = partials + (size_t)(bt * 16 + chunk) * 8;
        out[0] = m_p; out[1] = s_p; out[2] = m_n; out[3] = s_n;
        out[4] = cp;  out[5] = wp;  out[6] = cn;  out[7] = wn;
    }
}

// ---------------- kernel 3: final reduce + scalar loss ----------------
__global__ void __launch_bounds__(256)
circle_finalize(const float* __restrict__ partials, int nparts, float* __restrict__ out) {
    const int wave    = threadIdx.x >> 5;
    const unsigned ln = threadIdx.x & 31u;
    float m_p = NEG_BIG, s_p = 0.f, m_n = NEG_BIG, s_n = 0.f;
    float cp = 0.f, wp = 0.f, cn = 0.f, wn = 0.f;

    for (int b = threadIdx.x; b < nparts; b += 256) {
        const float* q = partials + (size_t)b * 8;
        lse_merge(m_p, s_p, q[0], q[1]);
        lse_merge(m_n, s_n, q[2], q[3]);
        cp += q[4]; wp += q[5]; cn += q[6]; wn += q[7];
    }
    #pragma unroll
    for (int off = 16; off > 0; off >>= 1) {
        lse_merge(m_p, s_p, __shfl_xor(m_p, off, 32), __shfl_xor(s_p, off, 32));
        lse_merge(m_n, s_n, __shfl_xor(m_n, off, 32), __shfl_xor(s_n, off, 32));
        cp += __shfl_xor(cp, off, 32);
        wp += __shfl_xor(wp, off, 32);
        cn += __shfl_xor(cn, off, 32);
        wn += __shfl_xor(wn, off, 32);
    }
    __shared__ float red[8][8];
    if (ln == 0) {
        float* q = red[wave];
        q[0] = m_p; q[1] = s_p; q[2] = m_n; q[3] = s_n;
        q[4] = cp;  q[5] = wp;  q[6] = cn;  q[7] = wn;
    }
    __syncthreads();
    if (threadIdx.x == 0) {
        #pragma unroll
        for (int w = 1; w < 8; ++w) {
            const float* q = red[w];
            lse_merge(m_p, s_p, q[0], q[1]);
            lse_merge(m_n, s_n, q[2], q[3]);
            cp += q[4]; wp += q[5]; cn += q[6]; wn += q[7];
        }
        const float lse_p = (s_p > 0.f) ? (m_p + logf(s_p)) : NEG_BIG;
        const float lse_n = (s_n > 0.f) ? (m_n + logf(s_n)) : NEG_BIG;
        const float wpm = wp / fmaxf(cp, 1.f);
        const float wnm = wn / fmaxf(cn, 1.f);
        // stable softplus: max(x,0) + log1p(exp(-|x|))
        const float sp_p = fmaxf(lse_p, 0.f) + log1pf(__expf(-fabsf(lse_p)));
        const float sp_n = fmaxf(lse_n, 0.f) + log1pf(__expf(-fabsf(lse_n)));
        out[0] = wpm * sp_p + wnm * sp_n;
    }
}

extern "C" void kernel_launch(void* const* d_in, const int* in_sizes, int n_in,
                              void* d_out, int out_size, void* d_ws, size_t ws_size,
                              hipStream_t stream) {
    (void)in_sizes; (void)n_in; (void)out_size; (void)ws_size;
    const float* sim      = (const float*)d_in[0];   // [4096, 4096] f32
    const float* clusters = (const float*)d_in[1];   // [4096, 64]   f32 one-hot

    _Float16* cl16   = (_Float16*)d_ws;                                  // 512 KB
    float* partials  = (float*)((char*)d_ws + (size_t)NTOT * CDIM * 2);  // 4096*8 f32

    circle_cvt_f16<<<(NTOT * CDIM / 4 + 255) / 256, 256, 0, stream>>>(
        clusters, cl16, NTOT * CDIM / 4);

    circle_main<<<dim3(256, 16, 1), 256, 0, stream>>>(sim, cl16, partials);

    circle_finalize<<<1, 256, 0, stream>>>(partials, 4096, (float*)d_out);
}
